// NumEncoder_4020089389665
// MI455X (gfx1250) — matrix operand order repaired and checked by reading, not compiled, not run on hardware
//
#include <hip/hip_runtime.h>

#define Bsz 128
#define Ssz 512
#define Nsz 32
#define Dsz 512
#define Hsz 2
#define DD  (Dsz * Dsz)

typedef __attribute__((ext_vector_type(16))) __bf16 v16bf;
typedef __attribute__((ext_vector_type(8)))  __bf16 v8bf;
typedef __attribute__((ext_vector_type(8)))  float  v8f;

// ---------- WMMA fragment helpers (layouts per CDNA5 ISA 7.12.2) ----------

// A fragment, 16x32 bf16, row-major source with leading dim `ld`.
// lanes 0-15: row=lane, K 0..7 (V0-3) and 16..23 (V4-7)
// lanes 16-31: row=lane-16, K 8..15 and 24..31
static __device__ __forceinline__ v16bf load_fragA(const __bf16* base, int ld) {
  const int lane = threadIdx.x & 31;
  const int half = lane >> 4;
  const int r    = lane & 15;
  const __bf16* p = base + r * ld + half * 8;
  union { v16bf v; v8bf h[2]; } u;
  u.h[0] = *(const v8bf*)(p);
  u.h[1] = *(const v8bf*)(p + 16);
  return u.v;
}

// B fragment, 32(K)x16(N) bf16, loaded from TRANSPOSED storage BT[N][K]
// (row-major, leading dim `ld`): lanes 0-15: col=lane, K 0..7 and 16..23;
// lanes 16-31: col=lane-16, K 8..15 and 24..31.
static __device__ __forceinline__ v16bf load_fragB_T(const __bf16* base, int ld) {
  const int lane = threadIdx.x & 31;
  const int half = lane >> 4;
  const int n    = lane & 15;
  const __bf16* p = base + n * ld + half * 8;
  union { v16bf v; v8bf h[2]; } u;
  u.h[0] = *(const v8bf*)(p);
  u.h[1] = *(const v8bf*)(p + 16);
  return u.v;
}

static __device__ __forceinline__ v8f wmma_bf16(v16bf a, v16bf b, v8f c) {
  return __builtin_amdgcn_wmma_f32_16x16x32_bf16(false, a, false, b, (short)0, c,
                                                 false, false);
}

// ---------- weight prep: convert f32 -> bf16, transpose, fold gate ----------
__global__ void k_prep_w(const float* __restrict__ w1a, const float* __restrict__ w1b,
                         const float* __restrict__ w2a, const float* __restrict__ w2b,
                         const float* __restrict__ wg,  const float* __restrict__ wo,
                         __bf16* __restrict__ WT) {
  const int h   = blockIdx.y >> 3;
  const int mat = blockIdx.y & 7;
  const int idx = blockIdx.x * 256 + threadIdx.x;   // 0..DD-1
  const int n = idx / Dsz, k = idx % Dsz;
  float v;
  switch (mat) {
    case 0: v = w1a[h * DD + k * Dsz + n]; break;
    case 1: v = w1b[h * DD + k * Dsz + n]; break;
    case 2: v = w2a[h * DD + k * Dsz + n]; break;
    case 3: v = w2b[h * DD + k * Dsz + n]; break;
    case 4: v = wg[h * 4 * DD + k * Dsz + n]
              + wg[h * 4 * DD + (2 * Dsz + k) * Dsz + n]
              + wg[h * 4 * DD + (3 * Dsz + k) * Dsz + n]; break;
    case 5: v = wg[h * 4 * DD + (Dsz + k) * Dsz + n]
              + wg[h * 4 * DD + (2 * Dsz + k) * Dsz + n]
              - wg[h * 4 * DD + (3 * Dsz + k) * Dsz + n]; break;
    case 6: v = wo[h * 2 * DD + k * Dsz + n]; break;
    default: v = wo[h * 2 * DD + (Dsz + k) * Dsz + n]; break;
  }
  WT[((size_t)(h * 8 + mat) * Dsz + n) * Dsz + k] = (__bf16)v;
}

// ---------- build row-normalized adjacency (bf16) ----------
__global__ void k_graph(const int* __restrict__ order, __bf16* __restrict__ ggt,
                        __bf16* __restrict__ glt) {
  const int b = blockIdx.x;
  const int n = threadIdx.x;            // 0..31 (one wave)
  const int* ob = order + b * Nsz;
  const int on = ob[n];
  float sgt = 0.f, slt = 0.f;
  for (int m = 0; m < Nsz; ++m) {
    const int om = ob[m];
    float egt, elt;
    if (m == n) { egt = 1.f; elt = 1.f; }
    else {
      const float pair = (on > 0 && om > 0) ? 1.f : 0.f;
      egt = (on > om) ? pair : 0.f;
      elt = (on > om) ? 0.f : pair;
    }
    sgt += egt; slt += elt;
  }
  const float rgt = 1.f / sgt, rlt = 1.f / slt;
  for (int m = 0; m < Nsz; ++m) {
    const int om = ob[m];
    float egt, elt;
    if (m == n) { egt = 1.f; elt = 1.f; }
    else {
      const float pair = (on > 0 && om > 0) ? 1.f : 0.f;
      egt = (on > om) ? pair : 0.f;
      elt = (on > om) ? 0.f : pair;
    }
    ggt[(b * Nsz + n) * Nsz + m] = (__bf16)(egt * rgt);
    glt[(b * Nsz + n) * Nsz + m] = (__bf16)(elt * rlt);
  }
}

// ---------- node init: f32 -> bf16 (normal + transposed layouts) ----------
__global__ void k_init_node(const float* __restrict__ ne, __bf16* __restrict__ node,
                            __bf16* __restrict__ nodeT) {
  const int i = blockIdx.x * 256 + threadIdx.x;     // B*N*D
  const int b = i / (Nsz * Dsz);
  const int r = i - b * Nsz * Dsz;
  const int n = r / Dsz, d = r % Dsz;
  const float v = ne[i];
  node[i] = (__bf16)v;
  nodeT[(b * Dsz + d) * Nsz + n] = (__bf16)v;
}

// ---------- graph aggregation y[b] = g[b](32x32) @ x[b](32xD) via WMMA ----------
__global__ void k_agg(const __bf16* __restrict__ g, const __bf16* __restrict__ xT,
                      __bf16* __restrict__ y) {
  const int wave = (blockIdx.x << 3) | (threadIdx.x >> 5);
  const int b = wave >> 5;              // batch
  const int j = wave & 31;              // 16-wide tile over D
  const int lane = threadIdx.x & 31;
  const int half = lane >> 4;
  const int nloc = lane & 15;
  const __bf16* gb = g + b * (Nsz * Nsz);
  v16bf a0 = load_fragA(gb, Nsz);                 // rows 0..15
  v16bf a1 = load_fragA(gb + 16 * Nsz, Nsz);      // rows 16..31
  v16bf bf = load_fragB_T(xT + ((size_t)b * Dsz + j * 16) * Nsz, Nsz);
  v8f c0 = {}, c1 = {};
  c0 = wmma_bf16(a0, bf, c0);
  c1 = wmma_bf16(a1, bf, c1);
  const int col = j * 16 + nloc;
  __bf16* yb = y + (size_t)b * Nsz * Dsz;
#pragma unroll
  for (int r = 0; r < 8; ++r) {
    const int m = r + (half << 3);
    yb[m * Dsz + col]        = (__bf16)c0[r];
    yb[(16 + m) * Dsz + col] = (__bf16)c1[r];
  }
}

// ---------- main GEMM: C = epi(A1@W1 [+ A2@W2] + bias) ----------
// A: [4096,512] bf16 row-major; W*T: [512,512] bf16 (transposed weights)
// EPI 0: ReLU (+ optional transposed copy CT as one b128/lane/tile)
// EPI 1: sigmoid-gate mix
template <int EPI, bool DUAL>
__global__ void k_gemm(const __bf16* __restrict__ A1, const __bf16* __restrict__ W1T,
                       const __bf16* __restrict__ A2, const __bf16* __restrict__ W2T,
                       const float* __restrict__ bias,
                       const __bf16* __restrict__ e1, const __bf16* __restrict__ e2,
                       __bf16* __restrict__ C, __bf16* __restrict__ CT) {
  const int wave  = (blockIdx.x << 3) | (threadIdx.x >> 5);
  const int tm    = wave >> 3;            // 0..255  (M/16)
  const int nBase = (wave & 7) << 6;      // 0..448  (N in 64-wide strips)
  const int lane  = threadIdx.x & 31;
  const int half  = lane >> 4;
  const int nloc  = lane & 15;
  const int rowA  = tm << 4;

  v8f acc[4] = {};
  for (int k0 = 0; k0 < Dsz; k0 += 32) {
    v16bf a = load_fragA(A1 + (size_t)rowA * Dsz + k0, Dsz);
#pragma unroll
    for (int t = 0; t < 4; ++t) {
      v16bf bfr = load_fragB_T(W1T + (size_t)(nBase + t * 16) * Dsz + k0, Dsz);
      acc[t] = wmma_bf16(a, bfr, acc[t]);
    }
    if constexpr (DUAL) {
      v16bf a2 = load_fragA(A2 + (size_t)rowA * Dsz + k0, Dsz);
#pragma unroll
      for (int t = 0; t < 4; ++t) {
        v16bf bfr = load_fragB_T(W2T + (size_t)(nBase + t * 16) * Dsz + k0, Dsz);
        acc[t] = wmma_bf16(a2, bfr, acc[t]);
      }
    }
    if (k0 + 64 < Dsz) __builtin_prefetch(A1 + (size_t)rowA * Dsz + k0 + 64, 0, 1);
  }

#pragma unroll
  for (int t = 0; t < 4; ++t) {
    const int col = nBase + t * 16 + nloc;
    const float bb = bias[col];
    if constexpr (EPI == 0) {
      union { v8bf v; __bf16 e[8]; } pk;
#pragma unroll
      for (int r = 0; r < 8; ++r) {
        const int row = rowA + r + (half << 3);
        float v = acc[t][r] + bb;
        v = v > 0.f ? v : 0.f;
        pk.e[r] = (__bf16)v;
        C[(size_t)row * Dsz + col] = pk.e[r];
      }
      if (CT) {
        // rows rowA+8*half .. +7 share row>>5; nn consecutive -> 16B store
        const int row0 = rowA + (half << 3);
        const int bb2 = row0 >> 5, nn = row0 & 31;
        *(v8bf*)(CT + ((size_t)bb2 * Dsz + col) * Nsz + nn) = pk.v;
      }
    } else {
#pragma unroll
      for (int r = 0; r < 8; ++r) {
        const int row = rowA + r + (half << 3);
        const float v = acc[t][r] + bb;
        const float gte = 1.f / (1.f + __expf(-v));
        const float f1 = (float)e1[(size_t)row * Dsz + col];
        const float f2 = (float)e2[(size_t)row * Dsz + col];
        C[(size_t)row * Dsz + col] = (__bf16)(gte * f1 + (1.f - gte) * f2);
      }
    }
  }
}

// ---------- epilogue kernels ----------
__global__ void k_num_emb(const float* __restrict__ ne, const __bf16* __restrict__ node,
                          float* __restrict__ out) {
  const int i = blockIdx.x * 256 + threadIdx.x;
  out[i] = ne[i] + (float)node[i];
}

__global__ void k_copy_enc(const float4* __restrict__ enc, float4* __restrict__ out) {
  const int i = blockIdx.x * 256 + threadIdx.x;
  out[i] = enc[i];
}

__global__ void k_scatter(const float* __restrict__ enc, const int* __restrict__ pos,
                          const __bf16* __restrict__ node, float* __restrict__ out) {
  const int b = blockIdx.x >> 5;
  const int n = blockIdx.x & 31;
  const int p = pos[b * Nsz + n];
  if (p < 0) return;
  const float* eb = enc + ((size_t)p * Bsz + b) * Dsz;
  float* ob = out + ((size_t)p * Bsz + b) * Dsz;
  const __bf16* nb = node + ((size_t)b * Nsz + n) * Dsz;
  for (int d = threadIdx.x; d < Dsz; d += 256)
    ob[d] = eb[d] + (float)nb[d];
}

__global__ void k_maxred(const float* __restrict__ gnn, float* __restrict__ out) {
  const int i = blockIdx.x * 256 + threadIdx.x;   // 0..B*D-1
  const int b = i / Dsz, d = i % Dsz;
  float m = -3.4e38f;
  for (int s = 0; s < Ssz; ++s)
    m = fmaxf(m, gnn[((size_t)s * Bsz + b) * Dsz + d]);
  out[i] = m;
}

// ---------- host ----------
extern "C" void kernel_launch(void* const* d_in, const int* in_sizes, int n_in,
                              void* d_out, int out_size, void* d_ws, size_t ws_size,
                              hipStream_t stream) {
  (void)in_sizes; (void)n_in; (void)out_size; (void)ws_size;
  const float* enc   = (const float*)d_in[0];
  const float* nenc  = (const float*)d_in[1];
  const int*   pos   = (const int*)d_in[2];
  const int*   order = (const int*)d_in[3];
  const float* w1a = (const float*)d_in[4];
  const float* b1a = (const float*)d_in[5];
  const float* w1b = (const float*)d_in[6];
  const float* b1b = (const float*)d_in[7];
  const float* w2a = (const float*)d_in[8];
  const float* b2a = (const float*)d_in[9];
  const float* w2b = (const float*)d_in[10];
  const float* b2b = (const float*)d_in[11];
  const float* wg  = (const float*)d_in[12];
  const float* bg  = (const float*)d_in[13];
  const float* wo  = (const float*)d_in[14];
  const float* bo  = (const float*)d_in[15];

  float* out      = (float*)d_out;
  float* out_gnn  = out;                                   // [S,B,D]
  float* out_num  = out + (size_t)Ssz * Bsz * Dsz;         // [B,N,D]
  float* out_prob = out_num + (size_t)Bsz * Nsz * Dsz;     // [B,D]

  __bf16* ws = (__bf16*)d_ws;
  size_t o = 0;
  __bf16* WT  = ws + o; o += (size_t)Hsz * 8 * DD;         // 8 MiB
  __bf16* ggt = ws + o; o += (size_t)Bsz * Nsz * Nsz;
  __bf16* glt = ws + o; o += (size_t)Bsz * Nsz * Nsz;
  const size_t ACT = (size_t)Bsz * Nsz * Dsz;
  __bf16* buf[7];
  for (int i = 0; i < 7; ++i) { buf[i] = ws + o; o += ACT; }
  __bf16 *node = buf[0], *nodeT = buf[1], *tb = buf[2],
         *xa = buf[3], *xaT = buf[4], *n1 = buf[5], *n2 = buf[6];

  k_prep_w<<<dim3(DD / 256, Hsz * 8), 256, 0, stream>>>(w1a, w1b, w2a, w2b, wg, wo, WT);
  k_graph<<<Bsz, Nsz, 0, stream>>>(order, ggt, glt);
  k_init_node<<<(Bsz * Nsz * Dsz) / 256, 256, 0, stream>>>(nenc, node, nodeT);

  for (int h = 0; h < Hsz; ++h) {
    const __bf16* W = WT + (size_t)h * 8 * DD;
    const float *B1a = b1a + h * Dsz, *B1b = b1b + h * Dsz,
                *B2a = b2a + h * Dsz, *B2b = b2b + h * Dsz,
                *Bg  = bg  + h * Dsz, *Bo  = bo  + h * Dsz;

    k_agg<<<512, 256, 0, stream>>>(ggt, nodeT, tb);
    k_gemm<0, false><<<256, 256, 0, stream>>>(tb, W + 0 * DD, nullptr, nullptr, B1a,
                                              nullptr, nullptr, xa, xaT);
    k_agg<<<512, 256, 0, stream>>>(ggt, xaT, tb);
    k_gemm<0, false><<<256, 256, 0, stream>>>(tb, W + 1 * DD, nullptr, nullptr, B1b,
                                              nullptr, nullptr, n1, nullptr);
    k_agg<<<512, 256, 0, stream>>>(glt, nodeT, tb);
    k_gemm<0, false><<<256, 256, 0, stream>>>(tb, W + 2 * DD, nullptr, nullptr, B2a,
                                              nullptr, nullptr, xa, xaT);
    k_agg<<<512, 256, 0, stream>>>(glt, xaT, tb);
    k_gemm<0, false><<<256, 256, 0, stream>>>(tb, W + 3 * DD, nullptr, nullptr, B2b,
                                              nullptr, nullptr, n2, nullptr);
    // gate = sigmoid(ni1@Wg1 + ni2@Wg2 + bg); node_info = gate*ni1+(1-gate)*ni2
    k_gemm<1, true><<<256, 256, 0, stream>>>(n1, W + 4 * DD, n2, W + 5 * DD, Bg,
                                             n1, n2, tb, nullptr);
    // node' = relu(node@Wo1 + node_info@Wo2 + bo)
    k_gemm<0, true><<<256, 256, 0, stream>>>(node, W + 6 * DD, tb, W + 7 * DD, Bo,
                                             nullptr, nullptr, xa, xaT);
    __bf16* t0 = node; __bf16* t1 = nodeT;
    node = xa; nodeT = xaT; xa = t0; xaT = t1;
  }

  k_num_emb<<<(Bsz * Nsz * Dsz) / 256, 256, 0, stream>>>(nenc, node, out_num);
  k_copy_enc<<<(Ssz * Bsz * Dsz / 4) / 256, 256, 0, stream>>>((const float4*)enc,
                                                              (float4*)out_gnn);
  k_scatter<<<Bsz * Nsz, 256, 0, stream>>>(enc, pos, node, out_gnn);
  k_maxred<<<(Bsz * Dsz) / 256, 256, 0, stream>>>(out_gnn, out_prob);
}
